// GATConvBlock_74603581932064
// MI455X (gfx1250) — matrix operand order repaired and checked by reading, not compiled, not run on hardware
//
#include <hip/hip_runtime.h>
#include <hip/hip_bf16.h>

#define NNODES 50000
#define NEDGES 800000
#define ETOT   (NEDGES + NNODES)   // edges + self loops
#define CH     128                 // channels at every stage
#define H1     8
#define H2     1
#define NEG_SLOPE 0.2f

typedef __attribute__((ext_vector_type(16))) __bf16 v16bf;
typedef __attribute__((ext_vector_type(8)))  float  v8f;

// ---------------------------------------------------------------- utilities

__global__ void fill_kernel(float* __restrict__ p, float v, int count) {
    int i = blockIdx.x * blockDim.x + threadIdx.x;
    int stride = gridDim.x * blockDim.x;
    for (; i < count; i += stride) p[i] = v;
}

__device__ __forceinline__ void atomicMaxFloat(float* addr, float val) {
    // classic bit trick: works for mixed-sign floats, init must be -FLT_MAX
    if (val >= 0.0f)
        atomicMax((int*)addr, __float_as_int(val));
    else
        atomicMin((unsigned int*)addr, (unsigned int)__float_as_int(val));
}

// ---------------------------------------------------------------- GEMM (WMMA)
// H[m,n] = sum_k X[m,k] * W[k,n]; M=NNODES (multiple of 16), K=N=128.
// One wave per 16-row strip; computes all 8 column tiles, 4 K-steps of 32.
// bf16 inputs, f32 accumulation via v_wmma_f32_16x16x32_bf16.
__global__ __launch_bounds__(32) void gemm_wmma_bf16(
    const float* __restrict__ X, const float* __restrict__ W,
    float* __restrict__ Hout)
{
    const int m0   = blockIdx.x * 16;
    const int lane = threadIdx.x;        // 0..31
    const int hlf  = lane >> 4;          // lane half: 0 or 1
    const int l16  = lane & 15;

    v8f acc[8];
    #pragma unroll
    for (int n = 0; n < 8; ++n)
        #pragma unroll
        for (int r = 0; r < 8; ++r) acc[n][r] = 0.0f;

    #pragma unroll
    for (int kt = 0; kt < 4; ++kt) {
        const int kbase = kt * 32;
        // ---- A tile 16x32 (ISA 7.12.2 16-bit A layout): row M = l16,
        // element j (VGPR v=j/2, pos p=j&1): K = (v<4 ? 2v+p : 16+2(v-4)+p) + 8*hlf
        v16bf A;
        const float* xrow = X + (size_t)(m0 + l16) * CH + kbase + hlf * 8;
        #pragma unroll
        for (int j = 0; j < 16; ++j) {
            int v = j >> 1, p = j & 1;
            int k = (v < 4) ? (v * 2 + p) : (16 + (v - 4) * 2 + p);
            A[j] = (__bf16)xrow[k];
        }
        #pragma unroll
        for (int nt = 0; nt < 8; ++nt) {
            // ---- B tile 32x16: col N = l16, element j: K = (j<8 ? j : 16+(j-8)) + 8*hlf
            v16bf B;
            const float* wcol = W + (size_t)(kbase + hlf * 8) * CH + nt * 16 + l16;
            #pragma unroll
            for (int j = 0; j < 16; ++j) B[j] = (__bf16)wcol[(size_t)(j < 8 ? j : j + 8) * CH];
            acc[nt] = __builtin_amdgcn_wmma_f32_16x16x32_bf16(
                false, A, false, B, (short)0, acc[nt], false, false);
        }
    }
    // ---- D layout: VGPR r -> row = r + 8*hlf, col = l16
    #pragma unroll
    for (int nt = 0; nt < 8; ++nt) {
        #pragma unroll
        for (int r = 0; r < 8; ++r) {
            int row = m0 + r + 8 * hlf;
            Hout[(size_t)row * CH + nt * 16 + l16] = acc[nt][r];
        }
    }
}

// ---------------------------------------------------------------- scores
// s_src[n,h] = dot(feat[n, h*C:(h+1)*C], a_src[h]);   C = CH/Hh
__global__ void scores_kernel(const float* __restrict__ feat,
                              const float* __restrict__ asrc,
                              const float* __restrict__ adst,
                              float* __restrict__ ssrc,
                              float* __restrict__ sdst,
                              int n, int Hh)
{
    int i = blockIdx.x * blockDim.x + threadIdx.x;
    if (i >= n) return;
    const int C = CH / Hh;
    const float* f = feat + (size_t)i * CH;
    for (int h = 0; h < Hh; ++h) {
        float s1 = 0.f, s2 = 0.f;
        for (int c = 0; c < C; ++c) {
            float fv = f[h * C + c];
            s1 += fv * asrc[h * C + c];
            s2 += fv * adst[h * C + c];
        }
        ssrc[(size_t)i * Hh + h] = s1;
        sdst[(size_t)i * Hh + h] = s2;
    }
}

// ---------------------------------------------------------------- edge pass 1: segment max
__global__ void edge_max_kernel(const int* __restrict__ ei,
                                const float* __restrict__ ssrc,
                                const float* __restrict__ sdst,
                                float* __restrict__ m, int Hh)
{
    int e = blockIdx.x * blockDim.x + threadIdx.x;
    if (e >= ETOT) return;
    int s, d;
    if (e < NEDGES) { s = ei[e]; d = ei[NEDGES + e]; }
    else            { s = d = e - NEDGES; }
    for (int h = 0; h < Hh; ++h) {
        float v = ssrc[(size_t)s * Hh + h] + sdst[(size_t)d * Hh + h];
        v = v > 0.0f ? v : NEG_SLOPE * v;
        atomicMaxFloat(&m[(size_t)d * Hh + h], v);
    }
}

// ---------------------------------------------------------------- edge pass 2:
// den[d,h] += exp(e-m);  agg[d,:] += exp(e-m) * feat[s,:]   (unnormalized)
// One 32-lane group per edge, 4 channels per lane (always within one head).
__global__ __launch_bounds__(256) void edge_agg_kernel(
    const int* __restrict__ ei,
    const float* __restrict__ ssrc, const float* __restrict__ sdst,
    const float* __restrict__ m,    const float* __restrict__ feat,
    float* __restrict__ den, float* __restrict__ agg, int Hh)
{
    int group = (blockIdx.x * blockDim.x + threadIdx.x) >> 5;
    int lane  = threadIdx.x & 31;
    if (group >= ETOT) return;
    int s, d;
    if (group < NEDGES) { s = ei[group]; d = ei[NEDGES + group]; }
    else                { s = d = group - NEDGES; }
    const int C  = CH / Hh;
    const int c0 = lane * 4;
    const int h  = c0 / C;
    float v = ssrc[(size_t)s * Hh + h] + sdst[(size_t)d * Hh + h];
    v = v > 0.0f ? v : NEG_SLOPE * v;
    float ex = __expf(v - m[(size_t)d * Hh + h]);   // <= 1
    if ((c0 % C) == 0) atomicAdd(&den[(size_t)d * Hh + h], ex);
    const float* fs = feat + (size_t)s * CH + c0;
    float*       ad = agg  + (size_t)d * CH + c0;
    atomicAdd(&ad[0], ex * fs[0]);
    atomicAdd(&ad[1], ex * fs[1]);
    atomicAdd(&ad[2], ex * fs[2]);
    atomicAdd(&ad[3], ex * fs[3]);
}

// ---------------------------------------------------------------- finalize:
// agg = agg/(den+1e-16) + bias, and accumulate column sums for BN.
__global__ __launch_bounds__(128) void finalize_stats_kernel(
    float* __restrict__ agg, const float* __restrict__ den,
    const float* __restrict__ bias,
    float* __restrict__ colsum, float* __restrict__ colsumsq, int Hh)
{
    const int j = threadIdx.x;       // column 0..127
    const int C = CH / Hh;
    const int h = j / C;
    float sum = 0.f, sumsq = 0.f;
    for (int r = blockIdx.x; r < NNODES; r += gridDim.x) {
        float v = agg[(size_t)r * CH + j] / (den[(size_t)r * Hh + h] + 1e-16f) + bias[j];
        agg[(size_t)r * CH + j] = v;
        sum += v; sumsq += v * v;
    }
    atomicAdd(&colsum[j], sum);
    atomicAdd(&colsumsq[j], sumsq);
}

// ---------------------------------------------------------------- BN (batch stats) + SELU, in place
__global__ void bn_selu_kernel(float* __restrict__ xv,
                               const float* __restrict__ colsum,
                               const float* __restrict__ colsumsq,
                               const float* __restrict__ gamma,
                               const float* __restrict__ beta)
{
    int idx = blockIdx.x * blockDim.x + threadIdx.x;
    if (idx >= NNODES * CH) return;
    int j = idx & (CH - 1);
    float invn = 1.0f / (float)NNODES;
    float mu  = colsum[j] * invn;
    float var = colsumsq[j] * invn - mu * mu;     // biased variance
    float v = (xv[idx] - mu) * rsqrtf(var + 1e-5f);
    v = gamma[j] * v + beta[j];
    const float a  = 1.6732632423543772f;
    const float sc = 1.0507009873554805f;
    xv[idx] = v > 0.0f ? sc * v : sc * a * (__expf(v) - 1.0f);
}

// ---------------------------------------------------------------- launch

extern "C" void kernel_launch(void* const* d_in, const int* in_sizes, int n_in,
                              void* d_out, int out_size, void* d_ws, size_t ws_size,
                              hipStream_t stream) {
    const float* x   = (const float*)d_in[0];
    const int*   ei  = (const int*)  d_in[1];   // [2,E]: row0=src, row1=dst
    // d_in[2] edge_attr unused (edge_dim=None)
    const float* W1  = (const float*)d_in[3];
    const float* as1 = (const float*)d_in[4];
    const float* ad1 = (const float*)d_in[5];
    const float* b1  = (const float*)d_in[6];
    const float* g1  = (const float*)d_in[7];
    const float* be1 = (const float*)d_in[8];
    const float* W2  = (const float*)d_in[9];
    const float* as2 = (const float*)d_in[10];
    const float* ad2 = (const float*)d_in[11];
    const float* b2  = (const float*)d_in[12];
    const float* g2  = (const float*)d_in[13];
    const float* be2 = (const float*)d_in[14];
    float* out = (float*)d_out;

    // ---- workspace layout (floats)
    float* ws = (float*)d_ws;
    size_t off = 0;
    float* hbuf  = ws + off; off += (size_t)NNODES * CH;  // GEMM outputs (h1, then h2)
    float* abuf  = ws + off; off += (size_t)NNODES * CH;  // layer-1 aggregate -> x2
    float* ssrc1 = ws + off; off += (size_t)NNODES * H1;
    float* sdst1 = ws + off; off += (size_t)NNODES * H1;
    float* m1    = ws + off; off += (size_t)NNODES * H1;
    float* den1  = ws + off; off += (size_t)NNODES * H1;
    float* ssrc2 = ws + off; off += (size_t)NNODES;
    float* sdst2 = ws + off; off += (size_t)NNODES;
    float* m2    = ws + off; off += (size_t)NNODES;
    float* den2  = ws + off; off += (size_t)NNODES;
    float* cs1   = ws + off; off += CH;
    float* cq1   = ws + off; off += CH;
    float* cs2   = ws + off; off += CH;
    float* cq2   = ws + off; off += CH;

    const float NEG_INF = -3.402823466e38f;
    // ---- init (accumulators / maxima / stats)
    fill_kernel<<<2048, 256, 0, stream>>>(m1,   NEG_INF, NNODES * H1);
    fill_kernel<<<2048, 256, 0, stream>>>(den1, 0.0f,    NNODES * H1);
    fill_kernel<<<2048, 256, 0, stream>>>(m2,   NEG_INF, NNODES);
    fill_kernel<<<2048, 256, 0, stream>>>(den2, 0.0f,    NNODES);
    fill_kernel<<<2048, 256, 0, stream>>>(abuf, 0.0f,    NNODES * CH);
    fill_kernel<<<2048, 256, 0, stream>>>(out,  0.0f,    NNODES * CH);
    fill_kernel<<<1,    256, 0, stream>>>(cs1,  0.0f,    CH);
    fill_kernel<<<1,    256, 0, stream>>>(cq1,  0.0f,    CH);
    fill_kernel<<<1,    256, 0, stream>>>(cs2,  0.0f,    CH);
    fill_kernel<<<1,    256, 0, stream>>>(cq2,  0.0f,    CH);

    const int edgeBlocks    = (ETOT + 255) / 256;
    const int edgeAggBlocks = (ETOT + 7) / 8;       // 8 edge-groups of 32 lanes per 256-thr block
    const int nodeBlocks    = (NNODES + 255) / 256;
    const int elemBlocks    = (NNODES * CH + 255) / 256;

    // =========== Layer 1 ===========
    gemm_wmma_bf16<<<NNODES / 16, 32, 0, stream>>>(x, W1, hbuf);
    scores_kernel<<<nodeBlocks, 256, 0, stream>>>(hbuf, as1, ad1, ssrc1, sdst1, NNODES, H1);
    edge_max_kernel<<<edgeBlocks, 256, 0, stream>>>(ei, ssrc1, sdst1, m1, H1);
    edge_agg_kernel<<<edgeAggBlocks, 256, 0, stream>>>(ei, ssrc1, sdst1, m1, hbuf, den1, abuf, H1);
    finalize_stats_kernel<<<512, 128, 0, stream>>>(abuf, den1, b1, cs1, cq1, H1);
    bn_selu_kernel<<<elemBlocks, 256, 0, stream>>>(abuf, cs1, cq1, g1, be1);

    // =========== Layer 2 ===========
    gemm_wmma_bf16<<<NNODES / 16, 32, 0, stream>>>(abuf, W2, hbuf);
    scores_kernel<<<nodeBlocks, 256, 0, stream>>>(hbuf, as2, ad2, ssrc2, sdst2, NNODES, H2);
    edge_max_kernel<<<edgeBlocks, 256, 0, stream>>>(ei, ssrc2, sdst2, m2, H2);
    edge_agg_kernel<<<edgeAggBlocks, 256, 0, stream>>>(ei, ssrc2, sdst2, m2, hbuf, den2, out, H2);
    finalize_stats_kernel<<<512, 128, 0, stream>>>(out, den2, b2, cs2, cq2, H2);
    bn_selu_kernel<<<elemBlocks, 256, 0, stream>>>(out, cs2, cq2, g2, be2);
}
